// MultiVeh_GAT_40218073760395
// MI455X (gfx1250) — matrix operand order repaired and checked by reading, not compiled, not run on hardware
//
#include <hip/hip_runtime.h>

// ---------------------------------------------------------------------------
// GAT on MI455X (gfx1250): flash-attention-style fused GAT layers using
// v_wmma_f32_16x16x32_f16. wave32 blocks (1 wave per workgroup).
//
// Fragment layouts (CDNA5 ISA 7.12.2, 16-bit, wave32):
//   A 16x32:  lane L holds row M=L%16; half h holds K = h + 8*(h>=8) + 8*(L>=16)
//   B 32x16:  lane L holds col N=L%16; half h holds K = h + 8*(h>=8) + 8*(L>=16)
//   C/D f32:  VGPR r, lane L -> M = r + 8*(L>=16), N = L%16
// Producers store Wh in B-fragment layout (32B contiguous per lane per tile)
// so the attention inner loop loads B operands as plain b128 pairs.
// ---------------------------------------------------------------------------

typedef __attribute__((ext_vector_type(16))) _Float16 v16h;
typedef __attribute__((ext_vector_type(8)))  float    v8f;

#define ALPHA   0.2f
#define NEGINF  (-9.0e15f)

__device__ __forceinline__ v8f wmma_f16(v16h a, v16h b, v8f c) {
  return __builtin_amdgcn_wmma_f32_16x16x32_f16(false, a, false, b, (short)0, c,
                                                false, false);
}

union I4 { int4 v; int a[4]; };

// ---------------------------------------------------------------------------
// Layer-1 projection: Wh1[head] = x @ W_heads[head]  (K=64, f16 WMMA)
// Emits: Wh1 in B-fragment layout, f1 = Wh@a_src, g1 = Wh@a_dst (f32).
// grid = head(4) * batch(16) * jblock(32 rows each, 32 blocks)
// ---------------------------------------------------------------------------
__global__ __launch_bounds__(32)
void gat_l1_proj(const float* __restrict__ x, const float* __restrict__ W_heads,
                 const float* __restrict__ a_heads,
                 _Float16* __restrict__ whfrag, float* __restrict__ f1,
                 float* __restrict__ g1) {
  const int bid  = blockIdx.x;
  const int jt   = bid & 31;          // 32-row block
  const int b    = (bid >> 5) & 15;
  const int head = bid >> 9;
  const int lane = threadIdx.x;
  const int lhi  = (lane >= 16) ? 1 : 0;
  const int lm   = lane & 15;
  const int i0   = jt * 32;

  const float* W     = W_heads + head * 64 * 64;   // [64][64] (fin, fout)
  const float* a_src = a_heads + head * 128;
  const float* a_dst = a_src + 64;

  __shared__ float wh_s[32][64];

  const v8f vzero = {0.f, 0.f, 0.f, 0.f, 0.f, 0.f, 0.f, 0.f};
  v8f acc[2][4];
#pragma unroll
  for (int mt = 0; mt < 2; ++mt)
#pragma unroll
    for (int nt = 0; nt < 4; ++nt) acc[mt][nt] = vzero;

#pragma unroll
  for (int ks = 0; ks < 2; ++ks) {
    const int k0 = ks * 32;
    // B fragments: W[k][n], k = k0 + (h&7) + 8*lhi + 16*(h>>3), n = nt*16 + lm
    v16h bf[4];
#pragma unroll
    for (int nt = 0; nt < 4; ++nt) {
#pragma unroll
      for (int h = 0; h < 16; ++h) {
        const int k = k0 + (h & 7) + 8 * lhi + 16 * (h >> 3);
        bf[nt][h] = (_Float16)W[k * 64 + nt * 16 + lm];
      }
    }
#pragma unroll
    for (int mt = 0; mt < 2; ++mt) {
      const int row = i0 + mt * 16 + lm;
      const float* xr = x + ((size_t)b * 1024 + row) * 64 + k0 + lhi * 8;
      v16h af;
#pragma unroll
      for (int h = 0; h < 8; ++h) af[h] = (_Float16)xr[h];        // K run 0..7
#pragma unroll
      for (int h = 0; h < 8; ++h) af[8 + h] = (_Float16)xr[16 + h]; // K run 16..23
#pragma unroll
      for (int nt = 0; nt < 4; ++nt) acc[mt][nt] = wmma_f16(af, bf[nt], acc[mt][nt]);
    }
  }

  // D layout -> LDS (f32)
#pragma unroll
  for (int mt = 0; mt < 2; ++mt)
#pragma unroll
    for (int nt = 0; nt < 4; ++nt)
#pragma unroll
      for (int r = 0; r < 8; ++r)
        wh_s[mt * 16 + r + 8 * lhi][nt * 16 + lm] = acc[mt][nt][r];
  __syncthreads();

  // Per-row attention coefficients f,g (one row per lane)
  float fv = 0.f, gv = 0.f;
  for (int k = 0; k < 64; ++k) {
    const float w = wh_s[lane][k];
    fv += w * a_src[k];
    gv += w * a_dst[k];
  }
  f1[((size_t)head * 16 + b) * 1024 + i0 + lane] = fv;
  g1[((size_t)head * 16 + b) * 1024 + i0 + lane] = gv;

  // Store Wh in B-fragment layout: block = ((head*16+b)*32 + jt)*4 + nt
#pragma unroll
  for (int nt = 0; nt < 4; ++nt) {
    v16h of;
#pragma unroll
    for (int h = 0; h < 16; ++h)
      of[h] = (_Float16)wh_s[(h & 7) + 8 * lhi + 16 * (h >> 3)][nt * 16 + lm];
    const size_t blk = (((size_t)head * 16 + b) * 32 + jt) * 4 + nt;
    ((v16h*)whfrag)[blk * 32 + lane] = of;
  }
}

// ---------------------------------------------------------------------------
// Layer-1 attention (all 4 heads fused, adj tile shared across heads):
// pass1: online softmax stats over j; pass2: P tiles in A-fragment registers,
// WMMA against Wh B-fragments, ELU, store h2 f16 (concat over heads).
// grid = batch(16) * rowblock(64 blocks of 16 rows)
// ---------------------------------------------------------------------------
__global__ __launch_bounds__(32)
void gat_l1_attn(const int* __restrict__ adj, const _Float16* __restrict__ whfrag,
                 const float* __restrict__ f1, const float* __restrict__ g1,
                 _Float16* __restrict__ h2) {
  const int bid  = blockIdx.x;
  const int it   = bid & 63;
  const int b    = bid >> 6;
  const int i0   = it * 16;
  const int lane = threadIdx.x;
  const int lhi  = (lane >= 16) ? 1 : 0;
  const int lm   = lane & 15;

  __shared__ float g_s[4][1024];
  __shared__ float f_s[4][16];
  __shared__ float m_s[4][16];
  __shared__ float r_s[4][16];

  // Stage g vectors (4 heads x 1024) in LDS
#pragma unroll
  for (int head = 0; head < 4; ++head) {
    const float* gp = g1 + ((size_t)head * 16 + b) * 1024;
#pragma unroll
    for (int q = 0; q < 8; ++q) {
      const int n4 = q * 128 + lane * 4;
      *(float4*)&g_s[head][n4] = *(const float4*)&gp[n4];
    }
  }
  // Stage f for the 16 rows (64 values)
#pragma unroll
  for (int t = lane; t < 64; t += 32) {
    const int head = t >> 4, r = t & 15;
    f_s[head][r] = f1[((size_t)head * 16 + b) * 1024 + i0 + r];
  }
  __syncthreads();

  // ---- pass 1: online softmax stats (lane pair shares a row, halves of j) --
  const int row   = lm;
  const int jbase = lhi * 512;
  const int* adjr = adj + ((size_t)b * 1024 + i0 + row) * 1024 + jbase;
  float m[4], s[4], fr[4];
#pragma unroll
  for (int h = 0; h < 4; ++h) { m[h] = -3.0e38f; s[h] = 0.f; fr[h] = f_s[h][row]; }

  for (int q = 0; q < 128; ++q) {
    I4 av; av.v = ((const int4*)adjr)[q];
    const int ja = jbase + q * 4;
#pragma unroll
    for (int e4 = 0; e4 < 4; ++e4) {
      const int a = av.a[e4];
      const int j = ja + e4;
#pragma unroll
      for (int h = 0; h < 4; ++h) {
        float e = fr[h] + g_s[h][j];
        e = (e > 0.f) ? e : ALPHA * e;
        e = (a > 0) ? e : NEGINF;
        const float nm = fmaxf(m[h], e);
        s[h] = s[h] * __expf(m[h] - nm) + __expf(e - nm);
        m[h] = nm;
      }
    }
  }
  // merge partner lane (xor 16) and publish stats
#pragma unroll
  for (int h = 0; h < 4; ++h) {
    const float mo = __shfl_xor(m[h], 16, 32);
    const float so = __shfl_xor(s[h], 16, 32);
    const float nm = fmaxf(m[h], mo);
    const float ss = s[h] * __expf(m[h] - nm) + so * __expf(mo - nm);
    if (lane < 16) { m_s[h][row] = nm; r_s[h][row] = 1.0f / ss; }
  }
  __syncthreads();

  // ---- pass 2: K loop, P tiles -> WMMA -----------------------------------
  const v8f vzero = {0.f, 0.f, 0.f, 0.f, 0.f, 0.f, 0.f, 0.f};
  v8f acc[4][4];
#pragma unroll
  for (int h = 0; h < 4; ++h)
#pragma unroll
    for (int nt = 0; nt < 4; ++nt) acc[h][nt] = vzero;

  for (int jt = 0; jt < 32; ++jt) {
    const int j0 = jt * 32;
    const int* ar = adj + ((size_t)b * 1024 + i0 + lm) * 1024 + j0 + lhi * 8;
    if (jt + 1 < 32) __builtin_prefetch(ar + 32, 0, 1);
    I4 a0, a1, a2, a3;
    a0.v = *(const int4*)(ar);        // K offsets (0..3)+8*lhi
    a1.v = *(const int4*)(ar + 4);    // (4..7)
    a2.v = *(const int4*)(ar + 16);   // (16..19)
    a3.v = *(const int4*)(ar + 20);   // (20..23)
    int aa[16];
#pragma unroll
    for (int e4 = 0; e4 < 4; ++e4) {
      aa[e4]      = a0.a[e4];
      aa[4 + e4]  = a1.a[e4];
      aa[8 + e4]  = a2.a[e4];
      aa[12 + e4] = a3.a[e4];
    }

#pragma unroll
    for (int head = 0; head < 4; ++head) {
      const float fr2  = f_s[head][lm];
      const float mrow = m_s[head][lm];
      const float rrow = r_s[head][lm];
      v16h af;
#pragma unroll
      for (int h = 0; h < 16; ++h) {
        const int j = j0 + 8 * lhi + h + ((h >= 8) ? 8 : 0);
        float e = fr2 + g_s[head][j];
        e = (e > 0.f) ? e : ALPHA * e;
        const float p = (aa[h] > 0) ? __expf(e - mrow) * rrow : 0.0f;
        af[h] = (_Float16)p;
      }
      const size_t base = (((size_t)head * 16 + b) * 32 + jt) * 4;
#pragma unroll
      for (int nt = 0; nt < 4; ++nt) {
        const v16h bf = ((const v16h*)whfrag)[(base + nt) * 32 + lane];
        acc[head][nt] = wmma_f16(af, bf, acc[head][nt]);
      }
    }
  }

  // ---- epilogue: ELU + store concatenated h2 (f16) -----------------------
#pragma unroll
  for (int head = 0; head < 4; ++head)
#pragma unroll
    for (int nt = 0; nt < 4; ++nt)
#pragma unroll
      for (int r = 0; r < 8; ++r) {
        float v = acc[head][nt][r];
        v = (v > 0.f) ? v : (__expf(v) - 1.0f);
        const int row_o = i0 + r + 8 * lhi;
        const int col   = head * 64 + nt * 16 + lm;
        h2[((size_t)b * 1024 + row_o) * 256 + col] = (_Float16)v;
      }
}

// ---------------------------------------------------------------------------
// Layer-2 projection: Wh2 = h2 @ W_out  (K=256 f16 WMMA, 32 outputs)
// grid = batch(16) * jblock(32 rows, 32 blocks)
// ---------------------------------------------------------------------------
__global__ __launch_bounds__(32)
void gat_l2_proj(const _Float16* __restrict__ h2, const float* __restrict__ W_out,
                 const float* __restrict__ a_out, _Float16* __restrict__ whfrag2,
                 float* __restrict__ f2, float* __restrict__ g2) {
  const int bid  = blockIdx.x;
  const int jt   = bid & 31;
  const int b    = bid >> 5;
  const int lane = threadIdx.x;
  const int lhi  = (lane >= 16) ? 1 : 0;
  const int lm   = lane & 15;
  const int i0   = jt * 32;

  __shared__ float wh_s[32][32];

  const v8f vzero = {0.f, 0.f, 0.f, 0.f, 0.f, 0.f, 0.f, 0.f};
  v8f acc[2][2];
#pragma unroll
  for (int mt = 0; mt < 2; ++mt)
#pragma unroll
    for (int nt = 0; nt < 2; ++nt) acc[mt][nt] = vzero;

#pragma unroll
  for (int ks = 0; ks < 8; ++ks) {
    const int k0 = ks * 32;
    v16h bf[2];
#pragma unroll
    for (int nt = 0; nt < 2; ++nt) {
#pragma unroll
      for (int h = 0; h < 16; ++h) {
        const int k = k0 + (h & 7) + 8 * lhi + 16 * (h >> 3);
        bf[nt][h] = (_Float16)W_out[k * 32 + nt * 16 + lm];
      }
    }
#pragma unroll
    for (int mt = 0; mt < 2; ++mt) {
      const int row = i0 + mt * 16 + lm;
      const _Float16* hr = h2 + ((size_t)b * 1024 + row) * 256 + k0 + lhi * 8;
      v16h af;
#pragma unroll
      for (int h = 0; h < 8; ++h) af[h] = hr[h];
#pragma unroll
      for (int h = 0; h < 8; ++h) af[8 + h] = hr[16 + h];
#pragma unroll
      for (int nt = 0; nt < 2; ++nt) acc[mt][nt] = wmma_f16(af, bf[nt], acc[mt][nt]);
    }
  }

#pragma unroll
  for (int mt = 0; mt < 2; ++mt)
#pragma unroll
    for (int nt = 0; nt < 2; ++nt)
#pragma unroll
      for (int r = 0; r < 8; ++r)
        wh_s[mt * 16 + r + 8 * lhi][nt * 16 + lm] = acc[mt][nt][r];
  __syncthreads();

  float fv = 0.f, gv = 0.f;
  for (int k = 0; k < 32; ++k) {
    const float w = wh_s[lane][k];
    fv += w * a_out[k];
    gv += w * a_out[32 + k];
  }
  f2[(size_t)b * 1024 + i0 + lane] = fv;
  g2[(size_t)b * 1024 + i0 + lane] = gv;

#pragma unroll
  for (int nt = 0; nt < 2; ++nt) {
    v16h of;
#pragma unroll
    for (int h = 0; h < 16; ++h)
      of[h] = (_Float16)wh_s[(h & 7) + 8 * lhi + 16 * (h >> 3)][nt * 16 + lm];
    const size_t blk = ((size_t)b * 32 + jt) * 2 + nt;
    ((v16h*)whfrag2)[blk * 32 + lane] = of;
  }
}

// ---------------------------------------------------------------------------
// Layer-2 attention (single head, nclass=32, no ELU, f32 output)
// grid = batch(16) * rowblock(64)
// ---------------------------------------------------------------------------
__global__ __launch_bounds__(32)
void gat_l2_attn(const int* __restrict__ adj, const _Float16* __restrict__ whfrag2,
                 const float* __restrict__ f2, const float* __restrict__ g2,
                 float* __restrict__ out) {
  const int bid  = blockIdx.x;
  const int it   = bid & 63;
  const int b    = bid >> 6;
  const int i0   = it * 16;
  const int lane = threadIdx.x;
  const int lhi  = (lane >= 16) ? 1 : 0;
  const int lm   = lane & 15;

  __shared__ float g_s[1024];
  __shared__ float f_s[16];
  __shared__ float m_s[16];
  __shared__ float r_s[16];

  {
    const float* gp = g2 + (size_t)b * 1024;
#pragma unroll
    for (int q = 0; q < 8; ++q) {
      const int n4 = q * 128 + lane * 4;
      *(float4*)&g_s[n4] = *(const float4*)&gp[n4];
    }
  }
  if (lane < 16) f_s[lane] = f2[(size_t)b * 1024 + i0 + lane];
  __syncthreads();

  const int row   = lm;
  const int jbase = lhi * 512;
  const int* adjr = adj + ((size_t)b * 1024 + i0 + row) * 1024 + jbase;
  float m = -3.0e38f, s = 0.f;
  const float fr = f_s[row];
  for (int q = 0; q < 128; ++q) {
    I4 av; av.v = ((const int4*)adjr)[q];
    const int ja = jbase + q * 4;
#pragma unroll
    for (int e4 = 0; e4 < 4; ++e4) {
      float e = fr + g_s[ja + e4];
      e = (e > 0.f) ? e : ALPHA * e;
      e = (av.a[e4] > 0) ? e : NEGINF;
      const float nm = fmaxf(m, e);
      s = s * __expf(m - nm) + __expf(e - nm);
      m = nm;
    }
  }
  {
    const float mo = __shfl_xor(m, 16, 32);
    const float so = __shfl_xor(s, 16, 32);
    const float nm = fmaxf(m, mo);
    const float ss = s * __expf(m - nm) + so * __expf(mo - nm);
    if (lane < 16) { m_s[row] = nm; r_s[row] = 1.0f / ss; }
  }
  __syncthreads();

  const v8f vzero = {0.f, 0.f, 0.f, 0.f, 0.f, 0.f, 0.f, 0.f};
  v8f acc[2] = {vzero, vzero};

  for (int jt = 0; jt < 32; ++jt) {
    const int j0 = jt * 32;
    const int* ar = adj + ((size_t)b * 1024 + i0 + lm) * 1024 + j0 + lhi * 8;
    if (jt + 1 < 32) __builtin_prefetch(ar + 32, 0, 1);
    I4 a0, a1, a2, a3;
    a0.v = *(const int4*)(ar);
    a1.v = *(const int4*)(ar + 4);
    a2.v = *(const int4*)(ar + 16);
    a3.v = *(const int4*)(ar + 20);
    int aa[16];
#pragma unroll
    for (int e4 = 0; e4 < 4; ++e4) {
      aa[e4]      = a0.a[e4];
      aa[4 + e4]  = a1.a[e4];
      aa[8 + e4]  = a2.a[e4];
      aa[12 + e4] = a3.a[e4];
    }
    const float fr2  = f_s[lm];
    const float mrow = m_s[lm];
    const float rrow = r_s[lm];
    v16h af;
#pragma unroll
    for (int h = 0; h < 16; ++h) {
      const int j = j0 + 8 * lhi + h + ((h >= 8) ? 8 : 0);
      float e = fr2 + g_s[j];
      e = (e > 0.f) ? e : ALPHA * e;
      const float p = (aa[h] > 0) ? __expf(e - mrow) * rrow : 0.0f;
      af[h] = (_Float16)p;
    }
    const size_t base = ((size_t)b * 32 + jt) * 2;
#pragma unroll
    for (int nt = 0; nt < 2; ++nt) {
      const v16h bf = ((const v16h*)whfrag2)[(base + nt) * 32 + lane];
      acc[nt] = wmma_f16(af, bf, acc[nt]);
    }
  }

#pragma unroll
  for (int nt = 0; nt < 2; ++nt)
#pragma unroll
    for (int r = 0; r < 8; ++r) {
      const int row_o = i0 + r + 8 * lhi;
      out[((size_t)b * 1024 + row_o) * 32 + nt * 16 + lm] = acc[nt][r];
    }
}

// ---------------------------------------------------------------------------
// Host launcher. Workspace layout (bytes):
//   [0,        8 MiB)  Wh1 fragments (f16)   4*16*32*4 blocks * 1 KiB
//   [8M,      +256K)   f1 (f32)
//   [8.25M,   +256K)   g1 (f32)
//   [8.5M,    +8 MiB)  h2 (f16, row-major [b][n][256])
//   [16.5M,   +1 MiB)  Wh2 fragments (f16)
//   [17.5M,   +64K)    f2 ; [+64K) g2         total ~17.66 MiB
// ---------------------------------------------------------------------------
extern "C" void kernel_launch(void* const* d_in, const int* in_sizes, int n_in,
                              void* d_out, int out_size, void* d_ws, size_t ws_size,
                              hipStream_t stream) {
  const float* x       = (const float*)d_in[0];
  const int*   adj     = (const int*)d_in[1];
  const float* W_heads = (const float*)d_in[2];
  const float* a_heads = (const float*)d_in[3];
  const float* W_out   = (const float*)d_in[4];
  const float* a_out   = (const float*)d_in[5];
  float*       out     = (float*)d_out;

  char* w = (char*)d_ws;
  const size_t MB = 1024 * 1024, KB = 1024;
  _Float16* whfrag1 = (_Float16*)(w);
  float*    f1      = (float*)(w + 8 * MB);
  float*    g1      = (float*)(w + 8 * MB + 256 * KB);
  _Float16* h2      = (_Float16*)(w + 8 * MB + 512 * KB);
  _Float16* whfrag2 = (_Float16*)(w + 16 * MB + 512 * KB);
  float*    f2      = (float*)(w + 17 * MB + 512 * KB);
  float*    g2      = (float*)(w + 17 * MB + 576 * KB);

  gat_l1_proj<<<dim3(4 * 16 * 32), dim3(32), 0, stream>>>(x, W_heads, a_heads,
                                                          whfrag1, f1, g1);
  gat_l1_attn<<<dim3(16 * 64), dim3(32), 0, stream>>>(adj, whfrag1, f1, g1, h2);
  gat_l2_proj<<<dim3(16 * 32), dim3(32), 0, stream>>>(h2, W_out, a_out,
                                                      whfrag2, f2, g2);
  gat_l2_attn<<<dim3(16 * 64), dim3(32), 0, stream>>>(adj, whfrag2, f2, g2, out);
}
